// GNN_71425306132767
// MI455X (gfx1250) — matrix operand order repaired and checked by reading, not compiled, not run on hardware
//
#include <hip/hip_runtime.h>
#include <hip/hip_bf16.h>
#include <math.h>

#define NL 3

typedef _Float16 v16h __attribute__((ext_vector_type(16)));
typedef float    v8f  __attribute__((ext_vector_type(8)));

// ---- order-preserving float <-> uint key (for integer atomic min/max) ----
__device__ __forceinline__ unsigned fkey(float f) {
    unsigned u = __float_as_uint(f);
    return u ^ ((unsigned)((int)u >> 31) | 0x80000000u);
}
__device__ __forceinline__ float fdecode(unsigned k) {
    unsigned u = (k & 0x80000000u) ? (k ^ 0x80000000u) : ~k;
    return __uint_as_float(u);
}

// Load one CDNA5 16-bit A-fragment row (lane k-group g) from a 32-float row:
// floats [16g .. 16g+15] (contiguous k order), as 4x b128 loads.
__device__ __forceinline__ v16h load_a_frag(const float* __restrict__ row, int g) {
    const float4* p = (const float4*)row;
    float4 a0 = p[4 * g], a1 = p[4 * g + 1], a2 = p[4 * g + 2], a3 = p[4 * g + 3];
    v16h a;
    a[0]  = (_Float16)a0.x; a[1]  = (_Float16)a0.y; a[2]  = (_Float16)a0.z; a[3]  = (_Float16)a0.w;
    a[4]  = (_Float16)a1.x; a[5]  = (_Float16)a1.y; a[6]  = (_Float16)a1.z; a[7]  = (_Float16)a1.w;
    a[8]  = (_Float16)a2.x; a[9]  = (_Float16)a2.y; a[10] = (_Float16)a2.z; a[11] = (_Float16)a2.w;
    a[12] = (_Float16)a3.x; a[13] = (_Float16)a3.y; a[14] = (_Float16)a3.z; a[15] = (_Float16)a3.w;
    return a;
}

// =====================  degree / avg(log(deg+1))  =====================
__global__ void pna_init_deg(float* __restrict__ deg, float* __restrict__ acc, int N) {
    int i = blockIdx.x * blockDim.x + threadIdx.x;
    if (i < N) deg[i] = 0.f;
    if (i == 0) acc[0] = 0.f;
}

__global__ void pna_deg(const int* __restrict__ dst, float* __restrict__ deg, int E) {
    long long stride = (long long)gridDim.x * blockDim.x;
    for (long long e = (long long)blockIdx.x * blockDim.x + threadIdx.x; e < E; e += stride)
        atomicAdd(&deg[dst[e]], 1.f);
}

__global__ void pna_avg(const float* __restrict__ deg, float* __restrict__ acc, int N) {
    int i = blockIdx.x * blockDim.x + threadIdx.x;
    float v = (i < N) ? logf(deg[i] + 1.f) : 0.f;
    for (int o = 16; o > 0; o >>= 1) v += __shfl_down(v, o, 32);
    if ((threadIdx.x & 31) == 0) atomicAdd(acc, v);
}

// =====================  per-layer stats init  =====================
__global__ void pna_init_stats(float* __restrict__ s, float* __restrict__ q,
                               unsigned* __restrict__ mn, unsigned* __restrict__ mx,
                               long long n) {
    long long stride = (long long)gridDim.x * blockDim.x;
    for (long long i = (long long)blockIdx.x * blockDim.x + threadIdx.x; i < n; i += stride) {
        s[i] = 0.f; q[i] = 0.f; mn[i] = 0xFFFFFFFFu; mx[i] = 0u;
    }
}

// =====================  pack W_post into CDNA5 B-fragment lane layout  =====================
// Bpack[((ks*2 + nt)*32 + lane)*16 + e] = W_post[(ks*32 + 16*(lane>>4) + e)*32 + nt*16 + (lane&15)]
__global__ void pna_pack_b(const float* __restrict__ Wpost, _Float16* __restrict__ Bpack) {
    int p = blockIdx.x * blockDim.x + threadIdx.x;
    if (p < 16 * 2 * 32 * 16) {
        int e = p & 15, lane = (p >> 4) & 31, nt = (p >> 9) & 1, ks = p >> 10;
        int k = ks * 32 + 16 * (lane >> 4) + e;
        int n = nt * 16 + (lane & 15);
        Bpack[p] = (_Float16)Wpost[k * 32 + n];
    }
}

// =====================  fused edge pre-NN (WMMA) + scatter aggregation  =====================
__global__ void __launch_bounds__(256) pna_edge(
    const float* __restrict__ xin, const int* __restrict__ src, const int* __restrict__ dst,
    const float* __restrict__ Wpre, const float* __restrict__ bpre,
    float* __restrict__ sumA, float* __restrict__ sqA,
    unsigned* __restrict__ mnA, unsigned* __restrict__ mxA,
    int E, int N)
{
    const int lane = threadIdx.x & 31;
    const int nl   = lane & 15;     // column within 16-wide tile (B/C/D layout)
    const int g    = lane >> 4;     // half-wave k-group

    // Preload B fragments: contiguous k order as in A -> lane holds col nl, K = kb + 16*g + e
    v16h B00, B01, B10, B11;
    for (int e = 0; e < 16; ++e) {
        int k0 = 16 * g + e;        // k-step 0 (features of x[dst])
        int k1 = 32 + 16 * g + e;   // k-step 1 (features of x[src])
        B00[e] = (_Float16)Wpre[k0 * 32 + nl];
        B01[e] = (_Float16)Wpre[k0 * 32 + 16 + nl];
        B10[e] = (_Float16)Wpre[k1 * 32 + nl];
        B11[e] = (_Float16)Wpre[k1 * 32 + 16 + nl];
    }
    const float bp0 = bpre[nl], bp1 = bpre[16 + nl];

    const long long nTiles = ((long long)E + 15) >> 4;
    long long wave = ((long long)blockIdx.x * blockDim.x + threadIdx.x) >> 5;
    const long long nWaves = ((long long)gridDim.x * blockDim.x) >> 5;

    for (long long t = wave; t < nTiles; t += nWaves) {
        const long long tb = t << 4;
        v16h aD, aS;
        int nodes[8];

        if (tb + 16 <= (long long)E) {
            // ---- fast path: wave-uniform, no per-lane predication ----
            const long long tn = t + nWaves;              // prefetch next tile's indices
            if (tn < nTiles) {
                __builtin_prefetch(dst + (tn << 4), 0, 0);
                __builtin_prefetch(src + (tn << 4), 0, 0);
            }
            const int ndn = dst[tb + nl];
            const int nsn = src[tb + nl];
            aD = load_a_frag(xin + (size_t)ndn * 32, g);
            aS = load_a_frag(xin + (size_t)nsn * 32, g);
            const int4* dp = (const int4*)(dst + tb + 8 * g);  // 16B-aligned
            int4 q0 = dp[0], q1 = dp[1];
            nodes[0] = q0.x; nodes[1] = q0.y; nodes[2] = q0.z; nodes[3] = q0.w;
            nodes[4] = q1.x; nodes[5] = q1.y; nodes[6] = q1.z; nodes[7] = q1.w;
        } else {
            // ---- cold ragged tail (unused when E % 16 == 0) ----
            const long long er = tb + nl;
            const bool ok = er < (long long)E;
            const long long erc = ok ? er : 0;
            const int ndn = dst[erc], nsn = src[erc];
            aD = load_a_frag(xin + (size_t)ndn * 32, g);
            aS = load_a_frag(xin + (size_t)nsn * 32, g);
            const _Float16 m = (_Float16)(ok ? 1.f : 0.f);
            for (int e = 0; e < 16; ++e) { aD[e] = aD[e] * m; aS[e] = aS[e] * m; }
            for (int r = 0; r < 8; ++r) {
                const long long e2 = tb + 8 * g + r;
                nodes[r] = (e2 < (long long)E) ? dst[e2] : N;  // row N = dummy sink
            }
        }

        v8f c0, c1;
        for (int i = 0; i < 8; ++i) { c0[i] = 0.f; c1[i] = 0.f; }
        c0 = __builtin_amdgcn_wmma_f32_16x16x32_f16(false, aD, false, B00, (short)0, c0, false, false);
        c0 = __builtin_amdgcn_wmma_f32_16x16x32_f16(false, aS, false, B10, (short)0, c0, false, false);
        c1 = __builtin_amdgcn_wmma_f32_16x16x32_f16(false, aD, false, B01, (short)0, c1, false, false);
        c1 = __builtin_amdgcn_wmma_f32_16x16x32_f16(false, aS, false, B11, (short)0, c1, false, false);

        // D layout: lane holds col nl(+16), rows m = r + 8*g  ->  scatter to dst node stats
        for (int r = 0; r < 8; ++r) {
            const int node = nodes[r];
            const float v0 = c0[r] + bp0;
            const float v1 = c1[r] + bp1;
            const size_t o0 = (size_t)node * 32 + nl;
            const size_t o1 = o0 + 16;
            atomicAdd(&sumA[o0], v0);      atomicAdd(&sqA[o0], v0 * v0);
            atomicMin(&mnA[o0], fkey(v0)); atomicMax(&mxA[o0], fkey(v0));
            atomicAdd(&sumA[o1], v1);      atomicAdd(&sqA[o1], v1 * v1);
            atomicMin(&mnA[o1], fkey(v1)); atomicMax(&mxA[o1], fkey(v1));
        }
    }
}

// =====================  node post-NN  =====================
// 32 nodes/block, 128 threads (4 waves = 2 m-tiles x 2 n-tiles), K=512 -> 16 wmma k-steps.
#define FSTRIDE 520   // 512 + 8 halfs pad (bank-conflict avoidance)
__global__ void __launch_bounds__(128) pna_node(
    const float* __restrict__ xin,
    const float* __restrict__ sumA, const float* __restrict__ sqA,
    const unsigned* __restrict__ mnA, const unsigned* __restrict__ mxA,
    const float* __restrict__ degA, const float* __restrict__ avgAcc,
    const _Float16* __restrict__ Bpack, const float* __restrict__ bpost,
    const float* __restrict__ Wlin,  const float* __restrict__ blin,
    float* __restrict__ xout, int N)
{
    __shared__ _Float16 featL[32 * FSTRIDE]; // 33,280 B
    __shared__ float    yL[32 * 33];         //  4,224 B
    const int tid  = threadIdx.x;
    const int base = blockIdx.x * 32;
    const float avg = avgAcc[0] / (float)N;

    // ---- stage [x | mean,sum,std,min,max | amp | att] as f16 ----
    for (int p = tid; p < 32 * 32; p += 128) {
        const int nn = p >> 5, d = p & 31;
        int node = base + nn; if (node >= N) node = N - 1;
        const size_t o = (size_t)node * 32 + d;
        const float deg  = degA[node];
        const float degc = fmaxf(deg, 1.f);
        const float inv  = 1.f / degc;
        const float s    = sumA[o];
        const float mean = s * inv;
        float var = sqA[o] * inv - mean * mean;
        const float sd = sqrtf((var > 0.f ? var : 0.f) + 1e-5f);
        const bool has = deg > 0.f;
        const float mn = has ? fdecode(mnA[o]) : 0.f;
        const float mx = has ? fdecode(mxA[o]) : 0.f;
        const float ldg = logf(degc + 1.f);
        const float amp = ldg / avg;
        const float att = avg / ldg;
        _Float16* row = &featL[nn * FSTRIDE];
        row[d] = (_Float16)xin[o];
        const float ag[5] = {mean, s, sd, mn, mx};
        for (int b = 0; b < 5; ++b) {
            const float a = ag[b];
            row[32  + b * 32 + d] = (_Float16)a;
            row[192 + b * 32 + d] = (_Float16)(a * amp);
            row[352 + b * 32 + d] = (_Float16)(a * att);
        }
    }
    __syncthreads();

    // ---- [32 x 512] x [512 x 32] via WMMA, A from LDS, B from packed fragments ----
    const int wv = tid >> 5, lane = tid & 31;
    const int nl = lane & 15, g = lane >> 4;
    const int mtile = wv & 1, ntile = wv >> 1;
    const int n = ntile * 16 + nl;
    v8f acc; for (int i = 0; i < 8; ++i) acc[i] = 0.f;
    const _Float16* arow = &featL[(mtile * 16 + nl) * FSTRIDE];
    for (int ks = 0; ks < 16; ++ks) {
        const int kb = ks * 32;
        v16h a, b;
        const _Float16* f1 = arow + kb + 16 * g;  // ds_load_b128
        const _Float16* f2 = f1 + 8;              // ds_load_b128
        for (int e = 0; e < 8; ++e) { a[e] = f1[e]; a[8 + e] = f2[e]; }
        const _Float16* bp = Bpack + (((size_t)(ks * 2 + ntile) * 32 + lane) << 4);
        for (int e = 0; e < 16; ++e) b[e] = bp[e]; // 2x global_load_b128
        acc = __builtin_amdgcn_wmma_f32_16x16x32_f16(false, a, false, b, (short)0, acc, false, false);
    }
    const float bb = bpost[n];
    for (int r = 0; r < 8; ++r) {
        const int m = mtile * 16 + r + 8 * g;
        yL[m * 33 + n] = acc[r] + bb;
    }
    __syncthreads();

    // ---- tiny 32x32 lin + relu (VALU; negligible FLOPs) ----
    for (int p = tid; p < 32 * 32; p += 128) {
        const int nn = p >> 5, d = p & 31;
        const int node = base + nn;
        if (node < N) {
            float v = blin[d];
            const float* yr = &yL[nn * 33];
            for (int k = 0; k < 32; ++k) v += yr[k] * Wlin[k * 32 + d];
            xout[(size_t)node * 32 + d] = fmaxf(v, 0.f);
        }
    }
}

// =====================  final projection  =====================
__global__ void pna_out(const float* __restrict__ x, const float* __restrict__ Wout,
                        const float* __restrict__ bout, float* __restrict__ out, int N) {
    int i = blockIdx.x * blockDim.x + threadIdx.x;
    if (i < N) {
        float s = bout[0];
        const float* r = x + (size_t)i * 32;
        for (int d = 0; d < 32; ++d) s += r[d] * Wout[d];
        out[i] = s;
    }
}

// =====================  host launcher  =====================
extern "C" void kernel_launch(void* const* d_in, const int* in_sizes, int n_in,
                              void* d_out, int out_size, void* d_ws, size_t ws_size,
                              hipStream_t stream)
{
    const float* x0    = (const float*)d_in[0];
    const int*   ei    = (const int*)  d_in[1];
    const float* Wpre  = (const float*)d_in[2];
    const float* bpre  = (const float*)d_in[3];
    const float* Wpost = (const float*)d_in[4];
    const float* bpost = (const float*)d_in[5];
    const float* Wlin  = (const float*)d_in[6];
    const float* blin  = (const float*)d_in[7];
    const float* Wout  = (const float*)d_in[8];
    const float* bout  = (const float*)d_in[9];
    const int N = in_sizes[0] / 32;
    const int E = in_sizes[1] / 2;
    const int* src = ei;
    const int* dst = ei + E;
    (void)n_in; (void)out_size; (void)ws_size;

    char* w = (char*)d_ws;
    size_t off = 0;
    auto alloc = [&](size_t bytes) -> void* {
        void* p = w + off; off += (bytes + 255) & ~(size_t)255; return p;
    };
    float*     xA     = (float*)    alloc((size_t)N * 32 * 4);
    float*     xB     = (float*)    alloc((size_t)N * 32 * 4);
    float*     sumA   = (float*)    alloc((size_t)(N + 1) * 32 * 4);
    float*     sqA    = (float*)    alloc((size_t)(N + 1) * 32 * 4);
    unsigned*  mnA    = (unsigned*) alloc((size_t)(N + 1) * 32 * 4);
    unsigned*  mxA    = (unsigned*) alloc((size_t)(N + 1) * 32 * 4);
    float*     degA   = (float*)    alloc((size_t)N * 4);
    float*     avgAcc = (float*)    alloc(256);
    _Float16*  Bpack  = (_Float16*) alloc((size_t)16 * 2 * 32 * 16 * 2);

    pna_init_deg<<<(N + 255) / 256, 256, 0, stream>>>(degA, avgAcc, N);
    pna_deg<<<1024, 256, 0, stream>>>(dst, degA, E);
    pna_avg<<<(N + 255) / 256, 256, 0, stream>>>(degA, avgAcc, N);

    const float* xin = x0;
    float* xout = xA;
    for (int l = 0; l < NL; ++l) {
        const long long sn = (long long)(N + 1) * 32;
        pna_init_stats<<<2048, 256, 0, stream>>>(sumA, sqA, mnA, mxA, sn);
        pna_edge<<<1024, 256, 0, stream>>>(xin, src, dst,
                                           Wpre + (size_t)l * 64 * 32, bpre + l * 32,
                                           sumA, sqA, mnA, mxA, E, N);
        pna_pack_b<<<(16 * 2 * 32 * 16 + 255) / 256, 256, 0, stream>>>(
            Wpost + (size_t)l * 512 * 32, Bpack);
        pna_node<<<(N + 31) / 32, 128, 0, stream>>>(xin, sumA, sqA, mnA, mxA, degA, avgAcc,
                                                    Bpack, bpost + l * 32,
                                                    Wlin + (size_t)l * 32 * 32, blin + l * 32,
                                                    xout, N);
        xin = xout;
        xout = (l == 0) ? xB : xA;
    }
    pna_out<<<(N + 255) / 256, 256, 0, stream>>>(xin, Wout, bout, (float*)d_out, N);
}
